// VGG_Small_24850680775267
// MI455X (gfx1250) — 8-bit matrix operand order repaired and checked by reading, compiled, not run on hardware
//
#include <hip/hip_runtime.h>
#include <cstdint>


typedef __attribute__((ext_vector_type(8))) int v8i;

static __device__ __forceinline__ int8_t sgn8(float v) {
  return v > 0.f ? (int8_t)1 : (v < 0.f ? (int8_t)-1 : (int8_t)0);
}

static __device__ __forceinline__ v8i b_from_int4(int4 lo, int4 hi4) {
  v8i r;
  r[0] = lo.x; r[1] = lo.y; r[2] = lo.z; r[3] = lo.w;
  r[4] = hi4.x; r[5] = hi4.y; r[6] = hi4.z; r[7] = hi4.w;
  return r;
}


__global__ void k_bin_conv_w(const float* __restrict__ w, int8_t* __restrict__ o,
                             int Cout, int Cin) {
  int idx = blockIdx.x * 256 + threadIdx.x;
  int total = Cout * Cin * 9;
  if (idx >= total) return;
  int ci = idx % Cin; int t = idx / Cin; int tap = t % 9; int co = t / 9;
  o[idx] = sgn8(w[(co * Cin + ci) * 9 + tap]);
}

__global__ void k_bin_fc_w(const float* __restrict__ w, int8_t* __restrict__ o, int n) {
  int idx = blockIdx.x * 256 + threadIdx.x;
  if (idx >= n) return;
  o[idx] = sgn8(w[idx]);
}


__global__ void k_conv1_bn_sign(const float* __restrict__ x, const float* __restrict__ w,
                                const float* __restrict__ bnp, int8_t* __restrict__ s1) {
  int idx = blockIdx.x * 256 + threadIdx.x;
  if (idx >= 128 * 34 * 34) return;
  int xp = idx % 34; int t = idx / 34; int yp = t % 34; int b = t / 34;
  int8_t* op = s1 + (size_t)idx * 128;
  if (xp == 0 || xp == 33 || yp == 0 || yp == 33) {
    for (int c = 0; c < 128; ++c) op[c] = (int8_t)-1;
    return;
  }
  int y = yp - 1, xx = xp - 1;
  float in[27];
  #pragma unroll
  for (int ci = 0; ci < 3; ++ci)
    #pragma unroll
    for (int ky = 0; ky < 3; ++ky)
      #pragma unroll
      for (int kx = 0; kx < 3; ++kx) {
        int iy = y + ky - 1, ix = xx + kx - 1;
        float v = 0.f;
        if (iy >= 0 && iy < 32 && ix >= 0 && ix < 32)
          v = x[((b * 3 + ci) * 32 + iy) * 32 + ix];
        in[ci * 9 + ky * 3 + kx] = v;
      }
  for (int co = 0; co < 128; ++co) {
    float acc = 0.f;
    #pragma unroll
    for (int q = 0; q < 27; ++q) acc += in[q] * w[co * 27 + q];
    float r = (acc - bnp[256 + co]) * rsqrtf(bnp[384 + co] + 1e-5f) * bnp[co] + bnp[128 + co];
    op[co] = sgn8(r);
  }
}


__global__ void __launch_bounds__(256)
k_bconv_wmma(const int8_t* __restrict__ act, const int8_t* __restrict__ wgt,
             int* __restrict__ out, int LH, int LW, int LCIN, int COUT) {
  __shared__ int8_t lds_a[32 * 512];
  const int H = 1 << LH, W = 1 << LW, CIN = 1 << LCIN;
  const int Hp = H + 2, Wp = W + 2;
  const int tid = threadIdx.x;
  const int lane = tid & 31;
  const int wave = tid >> 5;
  const int hi = (lane >> 4) & 1;
  const int l15 = lane & 15;
  const int m0row = blockIdx.y << 5;
  const int nbase = (blockIdx.x * 8 + wave) << 5;

  const int n0 = nbase + l15;
  const int n1 = n0 + 16;
  const int x0 = n0 & (W - 1), y0 = (n0 >> LW) & (H - 1), b0 = n0 >> (LW + LH);
  const int x1 = n1 & (W - 1), y1 = (n1 >> LW) & (H - 1), b1 = n1 >> (LW + LH);
  const int rb0 = ((b0 * Hp + y0) * Wp + x0) << LCIN;
  const int rb1 = ((b1 * Hp + y1) * Wp + x1) << LCIN;

  const unsigned* wgtu = (const unsigned*)wgt;
  const int rowDw = 1 << (LCIN - 2);
  const int stageDw = 32 << (LCIN - 2);

  v8i c00 = {0,0,0,0,0,0,0,0}, c01 = {0,0,0,0,0,0,0,0};
  v8i c10 = {0,0,0,0,0,0,0,0}, c11 = {0,0,0,0,0,0,0,0};

  for (int tap = 0; tap < 9; ++tap) {
    const int ky = tap / 3, kx = tap - ky * 3;
    __syncthreads();
    for (int i = tid; i < stageDw; i += 256) {
      const int m = i >> (LCIN - 2);
      const int d = i - (m << (LCIN - 2));
      ((unsigned*)lds_a)[i] = wgtu[((m0row + m) * 9 + tap) * rowDw + d];
    }
    __syncthreads();

    const int tapoff = ((ky * Wp + kx) << LCIN);
    for (int ci0 = 0; ci0 < CIN; ci0 += 64) {
      v8i a0, a1;
      #pragma unroll
      for (int q = 0; q < 4; ++q) {
        const int off = (l15 << LCIN) + ci0 + hi * 16 + (q >> 1) * 32 + (q & 1) * 8;
        int2 p0 = *(const int2*)(lds_a + off);
        int2 p1 = *(const int2*)(lds_a + off + (16 << LCIN));
        a0[2 * q] = p0.x; a0[2 * q + 1] = p0.y;
        a1[2 * q] = p1.x; a1[2 * q + 1] = p1.y;
      }
      const int8_t* pb0 = act + rb0 + tapoff + ci0 + hi * 16;
      const int8_t* pb1 = act + rb1 + tapoff + ci0 + hi * 16;
      v8i bv0 = b_from_int4(*(const int4*)pb0, *(const int4*)(pb0 + 32));
      v8i bv1 = b_from_int4(*(const int4*)pb1, *(const int4*)(pb1 + 32));
      if (ci0 + 64 < CIN) __builtin_prefetch(pb0 + 64, 0, 0);

      c00 = __builtin_amdgcn_wmma_i32_16x16x64_iu8(true, a0, true, bv0, c00, false, false);
      c01 = __builtin_amdgcn_wmma_i32_16x16x64_iu8(true, a0, true, bv1, c01, false, false);
      c10 = __builtin_amdgcn_wmma_i32_16x16x64_iu8(true, a1, true, bv0, c10, false, false);
      c11 = __builtin_amdgcn_wmma_i32_16x16x64_iu8(true, a1, true, bv1, c11, false, false);
    }
  }

  const int co = m0row + hi * 8;
  int* o00 = out + (size_t)n0 * COUT + co;
  int* o01 = out + (size_t)n1 * COUT + co;
  *(int4*)(o00)          = make_int4(c00[0], c00[1], c00[2], c00[3]);
  *(int4*)(o00 + 4)      = make_int4(c00[4], c00[5], c00[6], c00[7]);
  *(int4*)(o01)          = make_int4(c01[0], c01[1], c01[2], c01[3]);
  *(int4*)(o01 + 4)      = make_int4(c01[4], c01[5], c01[6], c01[7]);
  *(int4*)(o00 + 16)     = make_int4(c10[0], c10[1], c10[2], c10[3]);
  *(int4*)(o00 + 20)     = make_int4(c10[4], c10[5], c10[6], c10[7]);
  *(int4*)(o01 + 16)     = make_int4(c11[0], c11[1], c11[2], c11[3]);
  *(int4*)(o01 + 20)     = make_int4(c11[4], c11[5], c11[6], c11[7]);
}


__global__ void __launch_bounds__(128)
k_bfc_wmma(const int8_t* __restrict__ act, const int8_t* __restrict__ wgt,
           int* __restrict__ out, int K, int M) {
  __shared__ int8_t lds_a[32 * 256];
  const int tid = threadIdx.x;
  const int lane = tid & 31;
  const int wave = tid >> 5;
  const int hi = (lane >> 4) & 1;
  const int l15 = lane & 15;
  const int m0row = blockIdx.y << 5;
  const int n0 = (wave << 5) + l15;
  const int n1 = n0 + 16;

  const unsigned* wgtu = (const unsigned*)wgt;

  v8i c00 = {0,0,0,0,0,0,0,0}, c01 = {0,0,0,0,0,0,0,0};
  v8i c10 = {0,0,0,0,0,0,0,0}, c11 = {0,0,0,0,0,0,0,0};

  for (int k0 = 0; k0 < K; k0 += 256) {
    __syncthreads();
    for (int i = tid; i < 2048; i += 128) {
      const int m = i >> 6;
      const int d = i & 63;
      ((unsigned*)lds_a)[i] = wgtu[(((m0row + m) * K + k0) >> 2) + d];
    }
    __syncthreads();

    #pragma unroll
    for (int kk = 0; kk < 256; kk += 64) {
      v8i a0, a1;
      #pragma unroll
      for (int q = 0; q < 4; ++q) {
        const int off = (l15 << 8) + kk + hi * 16 + (q >> 1) * 32 + (q & 1) * 8;
        int2 p0 = *(const int2*)(lds_a + off);
        int2 p1 = *(const int2*)(lds_a + off + (16 << 8));
        a0[2 * q] = p0.x; a0[2 * q + 1] = p0.y;
        a1[2 * q] = p1.x; a1[2 * q + 1] = p1.y;
      }
      const int8_t* pb0 = act + n0 * K + k0 + kk + hi * 16;
      const int8_t* pb1 = act + n1 * K + k0 + kk + hi * 16;
      v8i bv0 = b_from_int4(*(const int4*)pb0, *(const int4*)(pb0 + 32));
      v8i bv1 = b_from_int4(*(const int4*)pb1, *(const int4*)(pb1 + 32));

      c00 = __builtin_amdgcn_wmma_i32_16x16x64_iu8(true, a0, true, bv0, c00, false, false);
      c01 = __builtin_amdgcn_wmma_i32_16x16x64_iu8(true, a0, true, bv1, c01, false, false);
      c10 = __builtin_amdgcn_wmma_i32_16x16x64_iu8(true, a1, true, bv0, c10, false, false);
      c11 = __builtin_amdgcn_wmma_i32_16x16x64_iu8(true, a1, true, bv1, c11, false, false);
    }
    if (k0 + 256 < K) __builtin_prefetch(act + n0 * K + k0 + 256, 0, 0);
  }

  const int co = m0row + hi * 8;
  int* o00 = out + (size_t)n0 * M + co;
  int* o01 = out + (size_t)n1 * M + co;
  *(int4*)(o00)      = make_int4(c00[0], c00[1], c00[2], c00[3]);
  *(int4*)(o00 + 4)  = make_int4(c00[4], c00[5], c00[6], c00[7]);
  *(int4*)(o01)      = make_int4(c01[0], c01[1], c01[2], c01[3]);
  *(int4*)(o01 + 4)  = make_int4(c01[4], c01[5], c01[6], c01[7]);
  *(int4*)(o00 + 16) = make_int4(c10[0], c10[1], c10[2], c10[3]);
  *(int4*)(o00 + 20) = make_int4(c10[4], c10[5], c10[6], c10[7]);
  *(int4*)(o01 + 16) = make_int4(c11[0], c11[1], c11[2], c11[3]);
  *(int4*)(o01 + 20) = make_int4(c11[4], c11[5], c11[6], c11[7]);
}


__global__ void k_bn_sign_pad(const int* __restrict__ conv, const float* __restrict__ bnp,
                              int8_t* __restrict__ outs, int B, int H, int W, int C) {
  int Hp = H + 2, Wp = W + 2;
  int idx = blockIdx.x * 256 + threadIdx.x;
  int total = B * Hp * Wp * C;
  if (idx >= total) return;
  int cidx = idx % C; int t = idx / C; int xp = t % Wp; t /= Wp; int yp = t % Hp; int b = t / Hp;
  int8_t v = -1;
  if (xp > 0 && xp < Wp - 1 && yp > 0 && yp < Hp - 1) {
    int f = conv[((b * H + (yp - 1)) * W + (xp - 1)) * C + cidx];
    float r = ((float)f - bnp[2 * C + cidx]) * rsqrtf(bnp[3 * C + cidx] + 1e-5f) * bnp[cidx]
              + bnp[C + cidx];
    v = sgn8(r);
  }
  outs[idx] = v;
}

__global__ void k_pool_bn_sign_pad(const int* __restrict__ conv, const float* __restrict__ bnp,
                                   int8_t* __restrict__ outs, int B, int H, int W, int C) {
  int Ho = H >> 1, Wo = W >> 1, Hp = Ho + 2, Wp = Wo + 2;
  int idx = blockIdx.x * 256 + threadIdx.x;
  int total = B * Hp * Wp * C;
  if (idx >= total) return;
  int cidx = idx % C; int t = idx / C; int xp = t % Wp; t /= Wp; int yp = t % Hp; int b = t / Hp;
  int8_t v = -1;
  if (xp > 0 && xp < Wp - 1 && yp > 0 && yp < Hp - 1) {
    int y = (yp - 1) << 1, x = (xp - 1) << 1;
    const int* p = conv + ((size_t)(b * H + y) * W + x) * C + cidx;
    int m0 = p[0], m1 = p[C], m2 = p[(size_t)W * C], m3 = p[(size_t)W * C + C];
    int m = m0 > m1 ? m0 : m1; int n2 = m2 > m3 ? m2 : m3; if (n2 > m) m = n2;
    float r = ((float)m - bnp[2 * C + cidx]) * rsqrtf(bnp[3 * C + cidx] + 1e-5f) * bnp[cidx]
              + bnp[C + cidx];
    v = sgn8(r);
  }
  outs[idx] = v;
}

__global__ void k_pool_bn_sign_fc(const int* __restrict__ conv, const float* __restrict__ bnp,
                                  int8_t* __restrict__ fcin) {
  int idx = blockIdx.x * 256 + threadIdx.x;
  if (idx >= 128 * 512 * 16) return;
  int x = idx & 3; int t = idx >> 2; int y = t & 3; t >>= 2; int c = t & 511; int b = t >> 9;
  int yy = y << 1, xx = x << 1;
  const int* p = conv + ((size_t)(b * 8 + yy) * 8 + xx) * 512 + c;
  int m0 = p[0], m1 = p[512], m2 = p[8 * 512], m3 = p[8 * 512 + 512];
  int m = m0 > m1 ? m0 : m1; int n2 = m2 > m3 ? m2 : m3; if (n2 > m) m = n2;
  float r = ((float)m - bnp[1024 + c]) * rsqrtf(bnp[1536 + c] + 1e-5f) * bnp[c] + bnp[512 + c];
  fcin[idx] = sgn8(r);
}

__global__ void k_fc_bn_sign(const int* __restrict__ in, const float* __restrict__ bnp,
                             int8_t* __restrict__ o, int Cm1, int total) {
  int idx = blockIdx.x * 256 + threadIdx.x;
  if (idx >= total) return;
  int c = idx & Cm1; int C = Cm1 + 1;
  float r = ((float)in[idx] - bnp[2 * C + c]) * rsqrtf(bnp[3 * C + c] + 1e-5f) * bnp[c] + bnp[C + c];
  o[idx] = sgn8(r);
}

__global__ void k_fc_bn_relu(const int* __restrict__ in, const float* __restrict__ bnp,
                             float* __restrict__ o, int Cm1, int total) {
  int idx = blockIdx.x * 256 + threadIdx.x;
  if (idx >= total) return;
  int c = idx & Cm1; int C = Cm1 + 1;
  float r = ((float)in[idx] - bnp[2 * C + c]) * rsqrtf(bnp[3 * C + c] + 1e-5f) * bnp[c] + bnp[C + c];
  o[idx] = r > 0.f ? r : 0.f;
}

__global__ void k_head(const float* __restrict__ h, const float* __restrict__ w,
                       const float* __restrict__ bias, float* __restrict__ out) {
  int idx = blockIdx.x * 256 + threadIdx.x;
  if (idx >= 1280) return;
  int o = idx % 10, b = idx / 10;
  float acc = bias[o];
  const float* hp = h + b * 1024;
  const float* wp = w + o * 1024;
  for (int k = 0; k < 1024; ++k) acc += hp[k] * wp[k];
  out[idx] = acc;
}


extern "C" void kernel_launch(void* const* d_in, const int* in_sizes, int n_in,
                              void* d_out, int out_size, void* d_ws, size_t ws_size,
                              hipStream_t stream) {
  const float* x    = (const float*)d_in[0];
  const float* w_c1 = (const float*)d_in[1];
  const float* bn1a = (const float*)d_in[2];
  const float* wb1  = (const float*)d_in[3];
  const float* bn1b = (const float*)d_in[4];
  const float* wb2a = (const float*)d_in[5];
  const float* bn2a = (const float*)d_in[6];
  const float* wb2b = (const float*)d_in[7];
  const float* bn2b = (const float*)d_in[8];
  const float* wb3a = (const float*)d_in[9];
  const float* bn3a = (const float*)d_in[10];
  const float* wb3b = (const float*)d_in[11];
  const float* bn3b = (const float*)d_in[12];
  const float* wf1  = (const float*)d_in[13];
  const float* bnf1 = (const float*)d_in[14];
  const float* wf2  = (const float*)d_in[15];
  const float* bnf2 = (const float*)d_in[16];
  const float* wout = (const float*)d_in[17];
  const float* bout = (const float*)d_in[18];
  float* out = (float*)d_out;

  char* ws = (char*)d_ws;
  int8_t* swb1  = (int8_t*)(ws + 0);
  int8_t* swb2a = (int8_t*)(ws + 147456);
  int8_t* swb2b = (int8_t*)(ws + 442368);
  int8_t* swb3a = (int8_t*)(ws + 1032192);
  int8_t* swb3b = (int8_t*)(ws + 2211840);
  int8_t* swf1  = (int8_t*)(ws + 4571136);
  int8_t* swf2  = (int8_t*)(ws + 12959744);
  int8_t* SIGN  = (int8_t*)(ws + 14680064);
  int*    CONV  = (int*)   (ws + 35651584);
  int*    FC1   = (int*)   (ws + 102760448);
  int8_t* SFC1  = (int8_t*)(ws + 103284736);
  int*    FC2   = (int*)   (ws + 103415808);
  float*  H2    = (float*) (ws + 103940096);

  auto gs = [](int n) { return (n + 255) / 256; };

  k_bin_conv_w<<<gs(128 * 128 * 9), 256, 0, stream>>>(wb1,  swb1,  128, 128);
  k_bin_conv_w<<<gs(256 * 128 * 9), 256, 0, stream>>>(wb2a, swb2a, 256, 128);
  k_bin_conv_w<<<gs(256 * 256 * 9), 256, 0, stream>>>(wb2b, swb2b, 256, 256);
  k_bin_conv_w<<<gs(512 * 256 * 9), 256, 0, stream>>>(wb3a, swb3a, 512, 256);
  k_bin_conv_w<<<gs(512 * 512 * 9), 256, 0, stream>>>(wb3b, swb3b, 512, 512);
  k_bin_fc_w<<<gs(1024 * 8192), 256, 0, stream>>>(wf1, swf1, 1024 * 8192);
  k_bin_fc_w<<<gs(1024 * 1024), 256, 0, stream>>>(wf2, swf2, 1024 * 1024);

  k_conv1_bn_sign<<<gs(128 * 34 * 34), 256, 0, stream>>>(x, w_c1, bn1a, SIGN);

  k_bconv_wmma<<<dim3(512, 4), 256, 0, stream>>>(SIGN, swb1, CONV, 5, 5, 7, 128);
  k_pool_bn_sign_pad<<<gs(128 * 18 * 18 * 128), 256, 0, stream>>>(CONV, bn1b, SIGN, 128, 32, 32, 128);

  k_bconv_wmma<<<dim3(128, 8), 256, 0, stream>>>(SIGN, swb2a, CONV, 4, 4, 7, 256);
  k_bn_sign_pad<<<gs(128 * 18 * 18 * 256), 256, 0, stream>>>(CONV, bn2a, SIGN, 128, 16, 16, 256);

  k_bconv_wmma<<<dim3(128, 8), 256, 0, stream>>>(SIGN, swb2b, CONV, 4, 4, 8, 256);
  k_pool_bn_sign_pad<<<gs(128 * 10 * 10 * 256), 256, 0, stream>>>(CONV, bn2b, SIGN, 128, 16, 16, 256);

  k_bconv_wmma<<<dim3(32, 16), 256, 0, stream>>>(SIGN, swb3a, CONV, 3, 3, 8, 512);
  k_bn_sign_pad<<<gs(128 * 10 * 10 * 512), 256, 0, stream>>>(CONV, bn3a, SIGN, 128, 8, 8, 512);

  k_bconv_wmma<<<dim3(32, 16), 256, 0, stream>>>(SIGN, swb3b, CONV, 3, 3, 9, 512);
  k_pool_bn_sign_fc<<<gs(128 * 512 * 16), 256, 0, stream>>>(CONV, bn3b, SIGN);

  k_bfc_wmma<<<dim3(1, 32), 128, 0, stream>>>(SIGN, swf1, FC1, 8192, 1024);
  k_fc_bn_sign<<<gs(128 * 1024), 256, 0, stream>>>(FC1, bnf1, SFC1, 1023, 128 * 1024);

  k_bfc_wmma<<<dim3(1, 32), 128, 0, stream>>>(SFC1, swf2, FC2, 1024, 1024);
  k_fc_bn_relu<<<gs(128 * 1024), 256, 0, stream>>>(FC2, bnf2, H2, 1023, 128 * 1024);

  k_head<<<gs(1280), 256, 0, stream>>>(H2, wout, bout, out);
}
